// CLIP_multiloss_24507083391380
// MI455X (gfx1250) — matrix operand order repaired and checked by reading, not compiled, not run on hardware
//
#include <hip/hip_runtime.h>
#include <math.h>

// ---------------------------------------------------------------------------
// CLIP multi-label smooth-L1 loss, fused for MI455X (gfx1250, wave32, WMMA).
//
//   V      = [labels | ones]            : [4096, 256]  (f16 copy staged in ws)
//   G      = V V^T                      : computed per 16x16 tile via
//                                          v_wmma_f32_16x16x32_f16 (f32 acc)
//   inv_i  = 1/sqrt(sum labels^2 + 128) : fp32, from labels directly
//   ml     = floor(G_ij * inv_i * inv_j)    (reciprocal-norms: no per-element
//                                            v_div_scale/div_fmas expansion)
//   loss   = (sum smoothL1(out0,ml) + sum smoothL1(out1,ml)) / (2*B*B)
//
// Memory floor: streaming out0+out1 (128 MB) once @ 23.3 TB/s ~ 5.6us.
// f16 WMMA keeps the 8.6 GFLOP Gram GEMM well under that floor; V (2 MB)
// lives in the 192 MB L2 so fragments load straight from global. Per-element
// epilogue is now ~7 VALU ops (2 mul + floor + 2x smooth-L1), keeping the
// fused pass memory-bound.
// ---------------------------------------------------------------------------

typedef __attribute__((ext_vector_type(16))) _Float16 v16h;
typedef __attribute__((ext_vector_type(8)))  _Float16 v8h;
typedef __attribute__((ext_vector_type(8)))  float    v8f;

#define BDIM 4096
#define LDIM 128
#define KDIM 256  // 2*LDIM

// workspace layout (bytes)
#define WS_VH_OFF      0                         // 4096*256 halfs = 2 MB
#define WS_NORM_OFF    (BDIM * KDIM * 2)         // 4096 floats   = 16 KB
#define WS_PART_OFF    (WS_NORM_OFF + BDIM * 4)  // 8192 floats   = 32 KB
#define NPART          (64 * 128)                // main-kernel block count

// ---- Kernel 0: build f16 [labels|ones] matrix + fp32 inverse row norms -----
__global__ __launch_bounds__(LDIM) void prep_kernel(
    const float* __restrict__ labels,
    _Float16* __restrict__ Vh,
    float* __restrict__ invn) {
  __shared__ float red[LDIM];
  const int r = blockIdx.x;
  const int t = threadIdx.x;

  float v = labels[(size_t)r * LDIM + t];
  Vh[(size_t)r * KDIM + t]        = (_Float16)v;
  Vh[(size_t)r * KDIM + LDIM + t] = (_Float16)1.0f;

  red[t] = v * v;
  __syncthreads();
#pragma unroll
  for (int s = LDIM / 2; s > 0; s >>= 1) {
    if (t < s) red[t] += red[t + s];
    __syncthreads();
  }
  // full-precision reciprocal norm, computed once per row (4096 total)
  if (t == 0) invn[r] = 1.0f / sqrtf(red[0] + (float)LDIM);
}

// ---- Kernel 1: fused Gram(WMMA) + floor + smooth-L1 + block reduction ------
// Block = 256 threads = 8 wave32s, arranged 2 (M) x 4 (N); each wave owns one
// 16x16 C tile -> block tile is 32 (M) x 64 (N). Grid = (4096/64, 4096/32).
__global__ __launch_bounds__(256) void gram_loss_kernel(
    const float* __restrict__ out0,
    const float* __restrict__ out1,
    const _Float16* __restrict__ Vh,
    const float* __restrict__ invn,
    float* __restrict__ partials) {
  const int lane = threadIdx.x & 31;
  const int wave = threadIdx.x >> 5;            // 0..7
  const int m0 = blockIdx.y * 32 + (wave >> 2) * 16;
  const int n0 = blockIdx.x * 64 + (wave & 3) * 16;

  const int lmod = lane & 15;                   // N / M position within tile
  const int lhi  = lane >> 4;                   // half-wave selector

  // 16-bit A-fragment (16x32): lane holds row M=lmod, two contiguous 8-half
  // runs at K = lhi*16 and K = lhi*16+8 within each 32-wide K chunk.
  const _Float16* arow = Vh + (size_t)(m0 + lmod) * KDIM + lhi * 16;
  // 16-bit B-fragment (32x16): lane holds column N=lmod (= a V row for G=VV^T),
  // one contiguous 16-half run at K = lhi*16.
  const _Float16* brow = Vh + (size_t)(n0 + lmod) * KDIM + lhi * 16;

  // Prefetch the outputs tiles we will consume in the epilogue while the
  // WMMA K-loop runs (emits global_prefetch_b8); high locality -> near cache.
  {
    size_t poff = (size_t)(m0 + lhi * 8) * BDIM + (n0 + lmod);
    __builtin_prefetch(out0 + poff, 0, 3);
    __builtin_prefetch(out1 + poff, 0, 3);
  }

  v8f c = {};
#pragma unroll
  for (int kk = 0; kk < KDIM; kk += 32) {
    v8h alo = *(const v8h*)(arow + kk);         // 16B aligned
    v8h ahi = *(const v8h*)(arow + kk + 8);     // 16B aligned
    v16h a;
#pragma unroll
    for (int i = 0; i < 8; ++i) { a[i] = alo[i]; a[i + 8] = ahi[i]; }
    v16h b = *(const v16h*)(brow + kk);         // 32B aligned

    // (neg_a, A, neg_b, B, c_mod, C, reuse_a, reuse_b)
    c = __builtin_amdgcn_wmma_f32_16x16x32_f16(
        false, a, false, b, (short)0, c, false, false);
  }

  // Epilogue: C layout -> VGPR r, lane l: M = r + 8*(l>>4), N = l&15.
  const int j  = n0 + lmod;
  const float invj = invn[j];
  float lsum = 0.0f;
#pragma unroll
  for (int r = 0; r < 8; ++r) {
    const int i  = m0 + r + lhi * 8;
    const float invi = invn[i];
    const float ml = floorf(c[r] * (invi * invj));   // mul+mul+floor, no div
    const size_t off = (size_t)i * BDIM + j;         // coalesced across N-lanes
    float d0 = fabsf(out0[off] - ml);
    float d1 = fabsf(out1[off] - ml);
    lsum += (d0 < 1.0f) ? 0.5f * d0 * d0 : d0 - 0.5f;
    lsum += (d1 < 1.0f) ? 0.5f * d1 * d1 : d1 - 0.5f;
  }

  // Deterministic fixed-order block reduction (no float atomics).
  __shared__ float red[256];
  red[threadIdx.x] = lsum;
  __syncthreads();
#pragma unroll
  for (int s = 128; s > 0; s >>= 1) {
    if (threadIdx.x < s) red[threadIdx.x] += red[threadIdx.x + s];
    __syncthreads();
  }
  if (threadIdx.x == 0)
    partials[blockIdx.y * gridDim.x + blockIdx.x] = red[0];
}

// ---- Kernel 2: deterministic final reduction + scaling ---------------------
__global__ __launch_bounds__(256) void final_reduce_kernel(
    const float* __restrict__ partials,
    float* __restrict__ out,
    int n) {
  __shared__ float red[256];
  float s = 0.0f;
  for (int i = threadIdx.x; i < n; i += 256) s += partials[i];
  red[threadIdx.x] = s;
  __syncthreads();
#pragma unroll
  for (int k = 128; k > 0; k >>= 1) {
    if (threadIdx.x < k) red[threadIdx.x] += red[threadIdx.x + k];
    __syncthreads();
  }
  if (threadIdx.x == 0)
    out[0] = red[0] / (2.0f * (float)BDIM * (float)BDIM);
}

extern "C" void kernel_launch(void* const* d_in, const int* in_sizes, int n_in,
                              void* d_out, int out_size, void* d_ws, size_t ws_size,
                              hipStream_t stream) {
  const float* out0   = (const float*)d_in[0];   // [4096,4096] f32
  const float* out1   = (const float*)d_in[1];   // [4096,4096] f32
  const float* labels = (const float*)d_in[2];   // [4096,128]  f32
  float* out = (float*)d_out;

  char* ws = (char*)d_ws;
  _Float16* Vh   = (_Float16*)(ws + WS_VH_OFF);
  float*    invn = (float*)(ws + WS_NORM_OFF);
  float*    parts = (float*)(ws + WS_PART_OFF);

  // 0) stage f16 [labels|ones] + fp32 inverse norms
  prep_kernel<<<BDIM, LDIM, 0, stream>>>(labels, Vh, invn);

  // 1) fused WMMA Gram + smooth-L1 partial sums
  dim3 grid(BDIM / 64, BDIM / 32);   // (64, 128) -> 8192 blocks
  gram_loss_kernel<<<grid, 256, 0, stream>>>(out0, out1, Vh, invn, parts);

  // 2) final deterministic reduction
  final_reduce_kernel<<<1, 256, 0, stream>>>(parts, out, NPART);
}
